// ScaledDotProductAttention_17952963297975
// MI455X (gfx1250) — compile-verified, matrix operand order checked by reading, not run on hardware
//
#include <hip/hip_runtime.h>

typedef __attribute__((ext_vector_type(16))) _Float16 v16h;
typedef __attribute__((ext_vector_type(8)))  _Float16 v8h;
typedef __attribute__((ext_vector_type(4)))  _Float16 v4h;
typedef __attribute__((ext_vector_type(8)))  float    v8f;

namespace {
constexpr int Bc = 4, Hc = 16, Sc = 2048, Dc = 64;
constexpr int PSTR = 2056;  // padded f16 LDS row stride: 4112B = 1028 dwords -> bank stride 4, conflict-free
constexpr size_t NQ = (size_t)Bc * Hc * Sc * Dc;  // elements in q/k/v
}

// K-index of element e (0..15) of a 16-bit WMMA A/B fragment for lane-half hf (ISA 7.12.2)
__device__ __forceinline__ int kmap(int e, int hf) {
  return (e < 8 ? e : e + 8) + 8 * hf;
}

// elements 0..7 = contiguous K-run at 8*hf, elements 8..15 = contiguous K-run at 16+8*hf
__device__ __forceinline__ v16h combine8(v8h lo, v8h hi) {
  v16h r;
#pragma unroll
  for (int e = 0; e < 8; ++e) { r[e] = lo[e]; r[e + 8] = hi[e]; }
  return r;
}

// ---------------- pre-pass 1: q/8 and k -> f16 (elementwise, float4) ----------------
__global__ __launch_bounds__(256) void cvt_qk_kernel(const float* __restrict__ q,
                                                     const float* __restrict__ k,
                                                     _Float16* __restrict__ qh,
                                                     _Float16* __restrict__ kh) {
  const size_t i = ((size_t)blockIdx.x * blockDim.x + threadIdx.x) * 4;
  const float4 qv = *(const float4*)(q + i);
  const float4 kv = *(const float4*)(k + i);
  v4h qo = {(_Float16)(qv.x * 0.125f), (_Float16)(qv.y * 0.125f),
            (_Float16)(qv.z * 0.125f), (_Float16)(qv.w * 0.125f)};
  v4h ko = {(_Float16)kv.x, (_Float16)kv.y, (_Float16)kv.z, (_Float16)kv.w};
  *(v4h*)(qh + i) = qo;
  *(v4h*)(kh + i) = ko;
}

// ---------------- pre-pass 2: vt[bh][d][key] = f16(v[bh][key][d]) ----------------
__global__ __launch_bounds__(256) void vtrans_kernel(const float* __restrict__ v,
                                                     _Float16* __restrict__ vt) {
  __shared__ float tile[32 * 65];  // 32 keys x 64 d, padded
  const int bh = blockIdx.y;
  const int key0 = blockIdx.x * 32;
  const float* src = v + ((size_t)bh * Sc + key0) * Dc;
#pragma unroll
  for (int i = 0; i < 8; ++i) {
    const int l = threadIdx.x + i * 256;  // 0..2047, coalesced along d
    tile[(l >> 6) * 65 + (l & 63)] = src[l];
  }
  __syncthreads();
  _Float16* dst = vt + (size_t)bh * Dc * Sc + key0;
#pragma unroll
  for (int i = 0; i < 8; ++i) {
    const int o = threadIdx.x + i * 256;
    const int d = o >> 5, key = o & 31;  // coalesced along key
    dst[(size_t)d * Sc + key] = (_Float16)tile[key * 65 + d];
  }
}

// ---------------- main kernel (fast path): all-f16 fragment loads ----------------
__global__ __launch_bounds__(256) void sdpa_wmma_f16_kernel(
    const _Float16* __restrict__ qh, const _Float16* __restrict__ kh,
    const _Float16* __restrict__ vt, const float* __restrict__ bias,
    const float* __restrict__ mask, float* __restrict__ out_o,
    float* __restrict__ out_attn) {
  __shared__ __align__(16) _Float16 p_lds[16 * PSTR];  // 65792 B; phase-reused via barriers
  float* red   = (float*)p_lds;   // [8 waves][16 rows]  (softmax reductions)
  float* opart = (float*)p_lds;   // [8 waves][16x64]    (O partials, after P reads done)

  const int bh  = blockIdx.y;
  const int b   = bh >> 4;
  const int h   = bh & 15;
  const int q0  = blockIdx.x * 16;
  const int lane = threadIdx.x & 31;
  const int wave = threadIdx.x >> 5;
  const int hf   = lane >> 4;
  const int l16  = lane & 15;

  // Q tile: two A-fragments (K = d 0..31, 32..63), already scaled by 1/8
  const _Float16* qrow = qh + ((size_t)bh * Sc + q0 + l16) * Dc;
  const v16h aq0 = combine8(*(const v8h*)(qrow + 8 * hf), *(const v8h*)(qrow + 16 + 8 * hf));
  const v16h aq1 = combine8(*(const v8h*)(qrow + 32 + 8 * hf), *(const v8h*)(qrow + 48 + 8 * hf));

  // ---- scores: this wave's 256 columns = 16 tiles of 16x16, kept in registers ----
  v8f c[16];
#pragma unroll
  for (int t = 0; t < 16; ++t) {
    const int n0 = wave * 256 + t * 16;
    const _Float16* krow = kh + ((size_t)bh * Sc + n0 + l16) * Dc;  // B: lane = column n
    const v16h bk0 = combine8(*(const v8h*)(krow + 8 * hf), *(const v8h*)(krow + 16 + 8 * hf));
    const v16h bk1 = combine8(*(const v8h*)(krow + 32 + 8 * hf), *(const v8h*)(krow + 48 + 8 * hf));
    v8f z = {0.f, 0.f, 0.f, 0.f, 0.f, 0.f, 0.f, 0.f};
    v8f s = __builtin_amdgcn_wmma_f32_16x16x32_f16(false, aq0, false, bk0, (short)0, z, false, false);
    s     = __builtin_amdgcn_wmma_f32_16x16x32_f16(false, aq1, false, bk1, (short)0, s, false, false);
    const float* bb = bias + ((size_t)h * Sc + q0) * Sc + n0 + l16;
    const float* mb = mask + ((size_t)b * Sc + q0) * Sc + n0 + l16;
#pragma unroll
    for (int e = 0; e < 8; ++e) {   // C layout: element e -> row e+8*hf, col l16
      const int row = e + 8 * hf;
      s[e] += bb[(size_t)row * Sc] + mb[(size_t)row * Sc];
    }
    c[t] = s;
  }

  // ---- row max: shfl within 16-lane groups, then cross-wave via LDS ----
  float rmax[8];
#pragma unroll
  for (int e = 0; e < 8; ++e) {
    float mx = c[0][e];
#pragma unroll
    for (int t = 1; t < 16; ++t) mx = fmaxf(mx, c[t][e]);
#pragma unroll
    for (int off = 1; off < 16; off <<= 1) mx = fmaxf(mx, __shfl_xor(mx, off, 32));
    rmax[e] = mx;
  }
  if (l16 == 0) {
#pragma unroll
    for (int e = 0; e < 8; ++e) red[wave * 16 + 8 * hf + e] = rmax[e];
  }
  __syncthreads();
#pragma unroll
  for (int e = 0; e < 8; ++e) {
    const int row = e + 8 * hf;
    float mx = red[row];
    for (int w = 1; w < 8; ++w) mx = fmaxf(mx, red[w * 16 + row]);
    rmax[e] = mx;
  }
  __syncthreads();

  // ---- exp + row sum ----
  float rinv[8];
#pragma unroll
  for (int e = 0; e < 8; ++e) {
    float sum = 0.f;
#pragma unroll
    for (int t = 0; t < 16; ++t) {
      const float p = __expf(c[t][e] - rmax[e]);
      c[t][e] = p;
      sum += p;
    }
#pragma unroll
    for (int off = 1; off < 16; off <<= 1) sum += __shfl_xor(sum, off, 32);
    rinv[e] = sum;
  }
  if (l16 == 0) {
#pragma unroll
    for (int e = 0; e < 8; ++e) red[wave * 16 + 8 * hf + e] = rinv[e];
  }
  __syncthreads();
#pragma unroll
  for (int e = 0; e < 8; ++e) {
    const int row = e + 8 * hf;
    float ssum = 0.f;
    for (int w = 0; w < 8; ++w) ssum += red[w * 16 + row];
    rinv[e] = 1.0f / ssum;
  }
  __syncthreads();

  // ---- normalize: write attn (fp32, global) + P (f16, LDS) ----
  float* attn = out_attn + ((size_t)bh * Sc + q0) * Sc;
#pragma unroll
  for (int t = 0; t < 16; ++t) {
    const int n0 = wave * 256 + t * 16;
#pragma unroll
    for (int e = 0; e < 8; ++e) {
      const int row = e + 8 * hf;
      const float p = c[t][e] * rinv[e];
      attn[(size_t)row * Sc + n0 + l16] = p;
      p_lds[row * PSTR + n0 + l16] = (_Float16)p;
    }
  }
  __syncthreads();

  // ---- O_partial = P[:, wave slice] @ V[wave slice, :] (8 K-blocks x 4 d-tiles) ----
  v8f acc[4];
  {
    v8f z = {0.f, 0.f, 0.f, 0.f, 0.f, 0.f, 0.f, 0.f};
#pragma unroll
    for (int j = 0; j < 4; ++j) acc[j] = z;
  }
#pragma unroll
  for (int kb = 0; kb < 8; ++kb) {
    const int key0 = wave * 256 + kb * 32;
    const _Float16* pr = &p_lds[l16 * PSTR + key0];       // A: lane = query row
    const v16h aP = combine8(*(const v8h*)(pr + 8 * hf), *(const v8h*)(pr + 16 + 8 * hf));
#pragma unroll
    for (int j = 0; j < 4; ++j) {
      const _Float16* vr = vt + ((size_t)bh * Dc + j * 16 + l16) * Sc + key0;  // B: lane = d col
      const v16h bV = combine8(*(const v8h*)(vr + 8 * hf), *(const v8h*)(vr + 16 + 8 * hf));
      acc[j] = __builtin_amdgcn_wmma_f32_16x16x32_f16(false, aP, false, bV, (short)0, acc[j], false, false);
    }
  }
  __syncthreads();  // all P reads done -> LDS reusable for O partials

  // ---- deterministic cross-wave O reduction through LDS ----
#pragma unroll
  for (int j = 0; j < 4; ++j) {
#pragma unroll
    for (int e = 0; e < 8; ++e) {
      const int row = e + 8 * hf;
      opart[wave * 1024 + row * 64 + j * 16 + l16] = acc[j][e];
    }
  }
  __syncthreads();
  float* ob = out_o + ((size_t)bh * Sc + q0) * Dc;
  for (int i = threadIdx.x; i < 1024; i += 256) {
    float s = 0.f;
    for (int w = 0; w < 8; ++w) s += opart[w * 1024 + i];
    ob[i] = s;
  }
}

// ---------------- fallback (no workspace): converts f32->f16 in-kernel ----------------
__global__ __launch_bounds__(256) void sdpa_wmma_fallback_kernel(
    const float* __restrict__ q, const float* __restrict__ kmat,
    const float* __restrict__ vmat, const float* __restrict__ bias,
    const float* __restrict__ mask, float* __restrict__ out_o,
    float* __restrict__ out_attn) {
  __shared__ __align__(16) _Float16 p_lds[16 * PSTR];
  float* red   = (float*)p_lds;
  float* opart = (float*)p_lds;

  const int bh  = blockIdx.y;
  const int b   = bh >> 4;
  const int h   = bh & 15;
  const int q0  = blockIdx.x * 16;
  const int lane = threadIdx.x & 31;
  const int wave = threadIdx.x >> 5;
  const int hf   = lane >> 4;
  const int l16  = lane & 15;

  v16h aq0, aq1;
  {
    const float* qb = q + ((size_t)bh * Sc + q0 + l16) * Dc;
#pragma unroll
    for (int e = 0; e < 16; ++e) {
      const int kk = kmap(e, hf);
      aq0[e] = (_Float16)(qb[kk] * 0.125f);
      aq1[e] = (_Float16)(qb[32 + kk] * 0.125f);
    }
  }

  v8f c[16];
#pragma unroll
  for (int t = 0; t < 16; ++t) {
    const int n0 = wave * 256 + t * 16;
    v16h bk0, bk1;
    const float* kb = kmat + ((size_t)bh * Sc + n0 + l16) * Dc;
#pragma unroll
    for (int e = 0; e < 16; ++e) {
      const int kk = kmap(e, hf);
      bk0[e] = (_Float16)kb[kk];
      bk1[e] = (_Float16)kb[32 + kk];
    }
    v8f z = {0.f, 0.f, 0.f, 0.f, 0.f, 0.f, 0.f, 0.f};
    v8f s = __builtin_amdgcn_wmma_f32_16x16x32_f16(false, aq0, false, bk0, (short)0, z, false, false);
    s     = __builtin_amdgcn_wmma_f32_16x16x32_f16(false, aq1, false, bk1, (short)0, s, false, false);
    const float* bb = bias + ((size_t)h * Sc + q0) * Sc + n0 + l16;
    const float* mb = mask + ((size_t)b * Sc + q0) * Sc + n0 + l16;
#pragma unroll
    for (int e = 0; e < 8; ++e) {
      const int row = e + 8 * hf;
      s[e] += bb[(size_t)row * Sc] + mb[(size_t)row * Sc];
    }
    c[t] = s;
  }

  float rmax[8];
#pragma unroll
  for (int e = 0; e < 8; ++e) {
    float mx = c[0][e];
#pragma unroll
    for (int t = 1; t < 16; ++t) mx = fmaxf(mx, c[t][e]);
#pragma unroll
    for (int off = 1; off < 16; off <<= 1) mx = fmaxf(mx, __shfl_xor(mx, off, 32));
    rmax[e] = mx;
  }
  if (l16 == 0) {
#pragma unroll
    for (int e = 0; e < 8; ++e) red[wave * 16 + 8 * hf + e] = rmax[e];
  }
  __syncthreads();
#pragma unroll
  for (int e = 0; e < 8; ++e) {
    const int row = e + 8 * hf;
    float mx = red[row];
    for (int w = 1; w < 8; ++w) mx = fmaxf(mx, red[w * 16 + row]);
    rmax[e] = mx;
  }
  __syncthreads();

  float rinv[8];
#pragma unroll
  for (int e = 0; e < 8; ++e) {
    float sum = 0.f;
#pragma unroll
    for (int t = 0; t < 16; ++t) {
      const float p = __expf(c[t][e] - rmax[e]);
      c[t][e] = p;
      sum += p;
    }
#pragma unroll
    for (int off = 1; off < 16; off <<= 1) sum += __shfl_xor(sum, off, 32);
    rinv[e] = sum;
  }
  if (l16 == 0) {
#pragma unroll
    for (int e = 0; e < 8; ++e) red[wave * 16 + 8 * hf + e] = rinv[e];
  }
  __syncthreads();
#pragma unroll
  for (int e = 0; e < 8; ++e) {
    const int row = e + 8 * hf;
    float ssum = 0.f;
    for (int w = 0; w < 8; ++w) ssum += red[w * 16 + row];
    rinv[e] = 1.0f / ssum;
  }
  __syncthreads();

  float* attn = out_attn + ((size_t)bh * Sc + q0) * Sc;
#pragma unroll
  for (int t = 0; t < 16; ++t) {
    const int n0 = wave * 256 + t * 16;
#pragma unroll
    for (int e = 0; e < 8; ++e) {
      const int row = e + 8 * hf;
      const float p = c[t][e] * rinv[e];
      attn[(size_t)row * Sc + n0 + l16] = p;
      p_lds[row * PSTR + n0 + l16] = (_Float16)p;
    }
  }
  __syncthreads();

  v8f acc[4];
  {
    v8f z = {0.f, 0.f, 0.f, 0.f, 0.f, 0.f, 0.f, 0.f};
#pragma unroll
    for (int j = 0; j < 4; ++j) acc[j] = z;
  }
#pragma unroll
  for (int kb = 0; kb < 8; ++kb) {
    const int key0 = wave * 256 + kb * 32;
    const _Float16* pr = &p_lds[l16 * PSTR + key0];
    const v16h aP = combine8(*(const v8h*)(pr + 8 * hf), *(const v8h*)(pr + 16 + 8 * hf));
#pragma unroll
    for (int j = 0; j < 4; ++j) {
      v16h bV;
      const float* vb = vmat + ((size_t)bh * Sc + key0) * Dc + j * 16 + l16;
#pragma unroll
      for (int e = 0; e < 16; ++e) bV[e] = (_Float16)vb[(size_t)kmap(e, hf) * Dc];
      acc[j] = __builtin_amdgcn_wmma_f32_16x16x32_f16(false, aP, false, bV, (short)0, acc[j], false, false);
    }
  }
  __syncthreads();

#pragma unroll
  for (int j = 0; j < 4; ++j) {
#pragma unroll
    for (int e = 0; e < 8; ++e) {
      const int row = e + 8 * hf;
      opart[wave * 1024 + row * 64 + j * 16 + l16] = acc[j][e];
    }
  }
  __syncthreads();
  float* ob = out_o + ((size_t)bh * Sc + q0) * Dc;
  for (int i = threadIdx.x; i < 1024; i += 256) {
    float s = 0.f;
    for (int w = 0; w < 8; ++w) s += opart[w * 1024 + i];
    ob[i] = s;
  }
}

extern "C" void kernel_launch(void* const* d_in, const int* in_sizes, int n_in,
                              void* d_out, int out_size, void* d_ws, size_t ws_size,
                              hipStream_t stream) {
  (void)in_sizes; (void)n_in; (void)out_size;
  const float* q    = (const float*)d_in[0];
  const float* kmat = (const float*)d_in[1];
  const float* vmat = (const float*)d_in[2];
  const float* bias = (const float*)d_in[3];
  const float* mask = (const float*)d_in[4];
  float* out_o    = (float*)d_out;
  float* out_attn = out_o + NQ;  // tuple: (output, attn)

  const dim3 grid(Sc / 16, Bc * Hc);  // 128 query tiles x 64 (b,h)
  if (ws_size >= NQ * 2 * 3) {
    _Float16* qh = (_Float16*)d_ws;
    _Float16* kh = qh + NQ;
    _Float16* vt = kh + NQ;
    cvt_qk_kernel<<<(int)(NQ / (4 * 256)), 256, 0, stream>>>(q, kmat, qh, kh);
    vtrans_kernel<<<dim3(Sc / 32, Bc * Hc), 256, 0, stream>>>(vmat, vt);
    sdpa_wmma_f16_kernel<<<grid, 256, 0, stream>>>(qh, kh, vt, bias, mask, out_o, out_attn);
  } else {
    sdpa_wmma_fallback_kernel<<<grid, 256, 0, stream>>>(q, kmat, vmat, bias, mask, out_o, out_attn);
  }
}
